// CausalAttention_21741124452726
// MI455X (gfx1250) — matrix operand order repaired and checked by reading, not compiled, not run on hardware
//
#include <hip/hip_runtime.h>

#define BB   4
#define TT   2048
#define DIN  1024
#define DD   1024
#define BT   (BB * TT)

typedef __attribute__((ext_vector_type(16))) __bf16 v16bf;
typedef __attribute__((ext_vector_type(8)))  float  v8f;
typedef __attribute__((ext_vector_type(4)))  int    v4i;

union BfVec { v16bf bf; v4i q[2]; };

__device__ __forceinline__ unsigned short f2bf(float f) {
    unsigned int x = __float_as_uint(f);
    x += 0x7FFFu + ((x >> 16) & 1u);           // round-to-nearest-even
    return (unsigned short)(x >> 16);
}

// A-operand (16x32, 16-bit): lane holds K=16*l16+0..15 contiguous
// -> one 32-byte chunk. `p` points at element (row, 16*l16).
__device__ __forceinline__ v16bf load_a16(const unsigned short* p) {
    BfVec v;
    v.q[0] = *(const v4i*)(p);
    v.q[1] = *(const v4i*)(p + 8);
    return v.bf;
}

// B-operand (32x16, 16-bit): lane holds K=16*l16+0..15 contiguous -> one 32B chunk.
__device__ __forceinline__ v16bf load_b16(const unsigned short* p) {
    BfVec v;
    v.q[0] = *(const v4i*)(p);
    v.q[1] = *(const v4i*)(p + 8);
    return v.bf;
}

#define WMMA_BF16(a, b, c) \
    __builtin_amdgcn_wmma_f32_16x16x32_bf16(false, (a), false, (b), (short)0, (c), false, false)

// ---------------- conversion kernels ----------------

__global__ void k_cvt_bf16(const float* __restrict__ in, unsigned short* __restrict__ out, int n) {
    int i = (blockIdx.x * blockDim.x + threadIdx.x) * 4;
    if (i + 3 < n) {
        float4 f = *(const float4*)(in + i);
        ushort4 o;
        o.x = f2bf(f.x); o.y = f2bf(f.y); o.z = f2bf(f.z); o.w = f2bf(f.w);
        *(ushort4*)(out + i) = o;
    }
}

// W [DIN, DD] fp32 row-major  ->  Wt [DD, DIN] bf16 row-major
__global__ void k_wt_bf16(const float* __restrict__ w, unsigned short* __restrict__ wt) {
    int idx = blockIdx.x * blockDim.x + threadIdx.x;   // over DIN*DD
    int i = idx / DD;          // d_in
    int o = idx % DD;          // d_out
    wt[o * DIN + i] = f2bf(w[idx]);
}

// ---------------- projection GEMM: out[BT, DD] = Xb[BT, DIN] * Wt^T ----------------
// one wave computes a 16x64 tile; copy-free ping-pong pipelined operand loads.
// transposed==1 stores V^T layout [b][DD][TT].

__global__ void __launch_bounds__(256) k_proj(const unsigned short* __restrict__ X,
                                              const unsigned short* __restrict__ Wt,
                                              unsigned short* __restrict__ out,
                                              int transposed) {
    const int lane = threadIdx.x & 31;
    const int w    = threadIdx.x >> 5;
    const int gid  = blockIdx.x * 8 + w;
    const int NC64 = DD / 64;                  // 16
    const int mt   = gid / NC64;               // 0..511
    const int nt   = gid % NC64;
    const int l16  = lane >> 4;
    const int ln   = lane & 15;

    v8f acc[4] = { {}, {}, {}, {} };
    const unsigned short* arow = X + (size_t)(mt * 16 + ln) * DIN + 16 * l16;
    const unsigned short* brow[4];
#pragma unroll
    for (int j = 0; j < 4; ++j)
        brow[j] = Wt + (size_t)(nt * 64 + j * 16 + ln) * DIN + 16 * l16;

    // ping-pong pipeline: two fixed register sets, no rotation copies
    v16bf a0, a1, b0[4], b1[4];
    a0 = load_a16(arow);
#pragma unroll
    for (int j = 0; j < 4; ++j) b0[j] = load_b16(brow[j]);

    const int NCH = DIN / 32;                  // 32 chunks (even)
    for (int c = 0; c < NCH; c += 2) {
        // stage 1 load (chunk c+1 always exists since NCH is even)
        a1 = load_a16(arow + (c + 1) * 32);
#pragma unroll
        for (int j = 0; j < 4; ++j) b1[j] = load_b16(brow[j] + (c + 1) * 32);
        // consume chunk c
#pragma unroll
        for (int j = 0; j < 4; ++j) acc[j] = WMMA_BF16(a0, b0[j], acc[j]);
        // stage 0 load (chunk c+2)
        if (c + 2 < NCH) {
            a0 = load_a16(arow + (c + 2) * 32);
#pragma unroll
            for (int j = 0; j < 4; ++j) b0[j] = load_b16(brow[j] + (c + 2) * 32);
        }
        // consume chunk c+1
#pragma unroll
        for (int j = 0; j < 4; ++j) acc[j] = WMMA_BF16(a1, b1[j], acc[j]);
    }

#pragma unroll
    for (int j = 0; j < 4; ++j) {
        const int col = nt * 64 + j * 16 + ln;
#pragma unroll
        for (int r = 0; r < 8; ++r) {
            const int m = mt * 16 + r + 8 * l16;       // global row in [0, BT)
            unsigned short v = f2bf(acc[j][r]);
            if (!transposed) {
                out[(size_t)m * DD + col] = v;
            } else {
                out[(size_t)(m / TT) * DD * TT + (size_t)col * TT + (m % TT)] = v;
            }
        }
    }
}

// ---------------- flash attention ----------------
// block = 256 threads (8 waves), one batch b, one 16-query tile.
// waves split the 1024 output dims (128 dims each).
// V slice is staged to LDS with CDNA5 async-to-LDS DMA, overlapped with
// the QK^T WMMAs + cross-wave reduce + softmax; ordered via ASYNCcnt/DScnt.

__global__ void __launch_bounds__(256) k_attn(const unsigned short* __restrict__ Q,
                                              const unsigned short* __restrict__ K,
                                              const unsigned short* __restrict__ Vt,
                                              float* __restrict__ out) {
    __shared__ float Spart[8][512];                       // per-wave partial 16x32 scores
    __shared__ float Sred[512];                           // reduced, scaled, masked scores
    __shared__ __align__(32) unsigned short Pb[512];      // P tile bf16, row-major 16x32
    __shared__ float mrow[16], lrow[16], crow[16];        // online-softmax state
    extern __shared__ unsigned char dynsmem[];            // V tile: [DD][32] bf16 = 64 KB

    unsigned short* Vtile = (unsigned short*)dynsmem;

    const int tid   = threadIdx.x;
    const int lane  = tid & 31;
    const int w     = tid >> 5;
    const int l16   = lane >> 4;
    const int ln    = lane & 15;
    const int b     = blockIdx.x >> 7;       // /128
    const int qt    = blockIdx.x & 127;
    const int qbase = qt * 16;
    const int wdb   = w * 128;               // wave's dim base

    if (tid < 16) { mrow[tid] = -__builtin_inff(); lrow[tid] = 0.f; }

    // this wave's Q slice as 4 A-operands (covers 128 dims)
    v16bf Aq[4];
    const unsigned short* qrow = Q + (size_t)(b * TT + qbase + ln) * DD + wdb;
#pragma unroll
    for (int c = 0; c < 4; ++c)
        Aq[c] = load_a16(qrow + c * 32 + 16 * l16);

    v8f acc[8] = { {}, {}, {}, {}, {}, {}, {}, {} };

    const unsigned char* vslab = (const unsigned char*)(Vt + (size_t)b * DD * TT);

    const int nkt = (qbase + 47) >> 5;       // key tiles of 32 covering keys <= qbase+15
    for (int kt = 0; kt < nkt; ++kt) {
        const int kbase = kt * 32;

        // --- issue async DMA of this wave's V slice (dims wdb..wdb+127, 32 keys) ---
        // make sure our previous-iteration LDS reads of Vtile have returned
        asm volatile("s_wait_dscnt 0x0" ::: "memory");
#pragma unroll
        for (int i = 0; i < 16; ++i) {
            const int g   = i * 32 + lane;           // granule id, 16B each
            const int d   = g >> 2;                  // 0..127 local dim
            const int o   = (g & 3) * 16;            // byte offset within 64B key-row
            const int dim = wdb + d;
            const unsigned char* gp = vslab + ((size_t)dim * TT + kbase) * 2 + o;
            unsigned lp = (unsigned)(uintptr_t)((unsigned char*)Vtile + dim * 64 + o);
            asm volatile("global_load_async_to_lds_b128 %0, %1, off"
                         :: "v"(lp), "v"((unsigned long long)(uintptr_t)gp)
                         : "memory");
        }

        // --- prefetch next key tile's K rows (global_prefetch_b8) ---
        if (kt + 1 < nkt) {
            const unsigned short* pk = K + (size_t)(b * TT + kbase + 32 + lane) * DD + wdb;
            __builtin_prefetch(pk, 0, 1);
            __builtin_prefetch(pk + 64, 0, 1);
        }

        // --- partial S = Q_slice * K_slice^T (16 queries x 32 keys) ---
        // ping-pong pipelined B-operand loads, no rotation copies
        v8f ps[2] = { {}, {} };
        const unsigned short* krow0 = K + (size_t)(b * TT + kbase + ln) * DD + wdb + 16 * l16;
        const unsigned short* krow1 = krow0 + (size_t)16 * DD;   // keys +16
        v16bf k0a = load_b16(krow0);
        v16bf k1a = load_b16(krow1);
        v16bf k0b = load_b16(krow0 + 32);
        v16bf k1b = load_b16(krow1 + 32);
        ps[0] = WMMA_BF16(Aq[0], k0a, ps[0]);
        ps[1] = WMMA_BF16(Aq[0], k1a, ps[1]);
        k0a = load_b16(krow0 + 64);
        k1a = load_b16(krow1 + 64);
        ps[0] = WMMA_BF16(Aq[1], k0b, ps[0]);
        ps[1] = WMMA_BF16(Aq[1], k1b, ps[1]);
        k0b = load_b16(krow0 + 96);
        k1b = load_b16(krow1 + 96);
        ps[0] = WMMA_BF16(Aq[2], k0a, ps[0]);
        ps[1] = WMMA_BF16(Aq[2], k1a, ps[1]);
        ps[0] = WMMA_BF16(Aq[3], k0b, ps[0]);
        ps[1] = WMMA_BF16(Aq[3], k1b, ps[1]);

        __syncthreads();
#pragma unroll
        for (int n = 0; n < 2; ++n)
#pragma unroll
            for (int r = 0; r < 8; ++r)
                Spart[w][(r + 8 * l16) * 32 + n * 16 + ln] = ps[n][r];
        __syncthreads();

        // --- cross-wave reduce + scale + causal mask ---
#pragma unroll
        for (int e = 0; e < 2; ++e) {
            const int idx = tid * 2 + e;
            float s = 0.f;
#pragma unroll
            for (int ww = 0; ww < 8; ++ww) s += Spart[ww][idx];
            s *= 0.03125f;                                   // 1/sqrt(1024)
            const int row = idx >> 5;
            const int key = kbase + (idx & 31);
            if (key > qbase + row) s = -__builtin_inff();
            Sred[idx] = s;
        }
        __syncthreads();

        // --- online softmax (wave 0, one lane per query row) ---
        if (tid < 16) {
            const int row = tid;
            float mold = mrow[row];
            float mx = mold;
#pragma unroll
            for (int j = 0; j < 32; ++j) mx = fmaxf(mx, Sred[row * 32 + j]);
            float corr = __expf(mold - mx);
            float ssum = 0.f;
#pragma unroll
            for (int j = 0; j < 32; ++j) {
                float p = __expf(Sred[row * 32 + j] - mx);
                ssum += p;
                Pb[row * 32 + j] = f2bf(p);
            }
            mrow[row] = mx;
            lrow[row] = lrow[row] * corr + ssum;
            crow[row] = corr;
        }
        __syncthreads();

        // --- rescale O, then O += P * V_slice (V from LDS, async-staged) ---
        float cs[8];
#pragma unroll
        for (int r = 0; r < 8; ++r) cs[r] = crow[r + 8 * l16];
        v16bf Ap = load_a16(Pb + ln * 32 + 16 * l16);

        // our wave's V DMA must have landed
        asm volatile("s_wait_asynccnt 0x0" ::: "memory");

        v16bf vv[2];
        vv[0] = load_b16(Vtile + (size_t)(wdb + ln) * 32 + 16 * l16);
#pragma unroll
        for (int j = 0; j < 8; ++j) {
            if (j + 1 < 8)
                vv[(j + 1) & 1] =
                    load_b16(Vtile + (size_t)(wdb + (j + 1) * 16 + ln) * 32 + 16 * l16);
#pragma unroll
            for (int r = 0; r < 8; ++r) acc[j][r] *= cs[r];
            acc[j] = WMMA_BF16(Ap, vv[j & 1], acc[j]);
        }
    }

    __syncthreads();
    float li[8];
#pragma unroll
    for (int r = 0; r < 8; ++r) li[r] = 1.f / lrow[r + 8 * l16];
#pragma unroll
    for (int j = 0; j < 8; ++j) {
        const int col = wdb + j * 16 + ln;
#pragma unroll
        for (int r = 0; r < 8; ++r) {
            const int m = qbase + r + 8 * l16;
            out[(size_t)(b * TT + m) * DD + col] = acc[j][r] * li[r];
        }
    }
}

// ---------------- host launch ----------------

extern "C" void kernel_launch(void* const* d_in, const int* in_sizes, int n_in,
                              void* d_out, int out_size, void* d_ws, size_t ws_size,
                              hipStream_t stream) {
    (void)in_sizes; (void)n_in; (void)out_size; (void)ws_size;
    const float* X  = (const float*)d_in[0];
    const float* Wq = (const float*)d_in[1];
    const float* Wk = (const float*)d_in[2];
    const float* Wv = (const float*)d_in[3];
    float* out = (float*)d_out;

    char* ws = (char*)d_ws;
    size_t off = 0;
    auto wsAlloc = [&](size_t bytes) {
        void* p = ws + off;
        off += (bytes + 255) & ~(size_t)255;
        return p;
    };
    unsigned short* Xb  = (unsigned short*)wsAlloc((size_t)BT * DIN * 2);
    unsigned short* Wqt = (unsigned short*)wsAlloc((size_t)DIN * DD * 2);
    unsigned short* Wkt = (unsigned short*)wsAlloc((size_t)DIN * DD * 2);
    unsigned short* Wvt = (unsigned short*)wsAlloc((size_t)DIN * DD * 2);
    unsigned short* Qb  = (unsigned short*)wsAlloc((size_t)BT * DD * 2);
    unsigned short* Kb  = (unsigned short*)wsAlloc((size_t)BT * DD * 2);
    unsigned short* Vtb = (unsigned short*)wsAlloc((size_t)BT * DD * 2);

    // fp32 -> bf16 conversions
    k_cvt_bf16<<<(BT * DIN) / 1024, 256, 0, stream>>>(X, Xb, BT * DIN);
    k_wt_bf16<<<(DIN * DD) / 256, 256, 0, stream>>>(Wq, Wqt);
    k_wt_bf16<<<(DIN * DD) / 256, 256, 0, stream>>>(Wk, Wkt);
    k_wt_bf16<<<(DIN * DD) / 256, 256, 0, stream>>>(Wv, Wvt);

    // projections (WMMA GEMM); V stored transposed for the PV B-operand
    const int tiles = (BT / 16) * (DD / 64);   // 8192 tiles, 8 waves per block
    k_proj<<<tiles / 8, 256, 0, stream>>>(Xb, Wqt, Qb, 0);
    k_proj<<<tiles / 8, 256, 0, stream>>>(Xb, Wkt, Kb, 0);
    k_proj<<<tiles / 8, 256, 0, stream>>>(Xb, Wvt, Vtb, 1);

    // flash attention (64 KB dynamic LDS for the async-staged V tile)
    k_attn<<<BB * (TT / 16), 256, 64 * 1024, stream>>>(Qb, Kb, Vtb, out);
}
